// MultiHeadAttention_28724741276053
// MI455X (gfx1250) — compile-verified, matrix operand order checked by reading, not run on hardware
//
#include <hip/hip_runtime.h>
#include <stdint.h>

// ---------------------------------------------------------------------------
// MHA forward for MI455X (gfx1250, wave32).
// Compute: V_WMMA_F32_16X16X32_BF16 (fp32 accumulate).
// Data movement (attention): GLOBAL_LOAD_ASYNC_TO_LDS_B128 double-buffered
// K/V staging shared by the 4 waves of a workgroup, S_WAIT_ASYNCCNT sync.
// B=4, S=2048, D=1024, H=16, HD=64.
// ---------------------------------------------------------------------------

typedef __attribute__((ext_vector_type(16))) __bf16          v16bf;
typedef __attribute__((ext_vector_type(8)))  float           v8f;
typedef __attribute__((ext_vector_type(8)))  unsigned short  u16x8;
typedef __attribute__((ext_vector_type(4)))  float           f32x4;
typedef __attribute__((ext_vector_type(4)))  unsigned short  u16x4;

union BFrag {
  struct { u16x8 lo; u16x8 hi; } h;
  v16bf v;
};

__device__ __forceinline__ unsigned short f2bf(float x) {
  unsigned int u = __builtin_bit_cast(unsigned int, x);
  u += 0x7FFFu + ((u >> 16) & 1u);           // round-to-nearest-even
  return (unsigned short)(u >> 16);
}

// 16-bit A-operand fragment, 16x32 (MxK), per ISA 7.12.2:
//   lane m = lane&15; lanes 0-15 hold K[0..7] (v0-3) + K[16..23] (v4-7),
//   lanes 16-31 hold K[8..15] + K[24..31].
// B-operand (32x16, KxN) mirror layout with n = lane&15 — a matrix stored
// N-major (row = n, contiguous in k) loads identically. Works for both
// global and LDS pointers (LDS -> ds_load_b128).
__device__ __forceinline__ v16bf load_frag(const unsigned short* base, int ld, int lane) {
  const int r  = lane & 15;
  const int kb = (lane >> 4) << 3;           // 0 or 8
  const unsigned short* p = base + (size_t)r * ld;
  BFrag f;
  f.h.lo = *(const u16x8*)(p + kb);
  f.h.hi = *(const u16x8*)(p + 16 + kb);
  return f.v;
}

#define WMMA_BF16(A, B, C) \
  __builtin_amdgcn_wmma_f32_16x16x32_bf16(false, (A), false, (B), (short)0, (C), false, false)

// Async 16-byte global->LDS copy (per enabled lane). VDST carries the LDS
// byte offset (low 32 bits of a generic LDS pointer), ADDR the 64-bit
// global address. Tracked by ASYNCcnt.
__device__ __forceinline__ void async_copy_b128(unsigned lds_off, const void* gptr) {
  asm volatile("global_load_async_to_lds_b128 %0, %1, off"
               :: "v"(lds_off), "v"((unsigned long long)(uintptr_t)gptr)
               : "memory");
}

__device__ __forceinline__ void wait_async0() {
  asm volatile("s_wait_asynccnt 0x0" ::: "memory");
}

// ---------------------------------------------------------------------------
// Vectorized fp32 -> bf16 (4 elements/thread)
// ---------------------------------------------------------------------------
__global__ void __launch_bounds__(256)
f32_to_bf16x4_kernel(const float* __restrict__ in, unsigned short* __restrict__ out, int n4) {
  int i = blockIdx.x * 256 + threadIdx.x;
  if (i < n4) {
    f32x4 v = *(const f32x4*)(in + (size_t)i * 4);
    u16x4 o;
    o[0] = f2bf(v[0]); o[1] = f2bf(v[1]); o[2] = f2bf(v[2]); o[3] = f2bf(v[3]);
    *(u16x4*)(out + (size_t)i * 4) = o;
  }
}

// Tiled transpose + bf16 convert: out[n*K + k] = bf16(in[k*N + n]).
// 32x32 tile via LDS (padded to 33 cols), both sides coalesced.
// Grid: (K/32, N/32), block 256 (=32x8).
__global__ void __launch_bounds__(256)
transpose_bf16_kernel(const float* __restrict__ in, unsigned short* __restrict__ out,
                      int N, int K) {
  __shared__ float tile[32][33];
  const int tx = threadIdx.x & 31;
  const int ty = threadIdx.x >> 5;          // 0..7
  const int k0 = blockIdx.x * 32;
  const int n0 = blockIdx.y * 32;
#pragma unroll
  for (int i = 0; i < 4; ++i) {
    const int k = k0 + ty + i * 8;
    tile[ty + i * 8][tx] = in[(size_t)k * N + n0 + tx];
  }
  __syncthreads();
#pragma unroll
  for (int i = 0; i < 4; ++i) {
    const int n = n0 + ty + i * 8;
    out[(size_t)n * K + k0 + tx] = f2bf(tile[tx][ty + i * 8]);
  }
}

// ---------------------------------------------------------------------------
// GEMM helpers: one wave computes a 16x64 tile. A K-slice (depth 32) is one
// A fragment + four B fragments. The slice loop is FULLY unrolled: with
// straight-line SSA there are no loop-carried phis, so the prefetched
// fragments keep their own registers and no cur<-next copies are emitted;
// the scheduler staggers the 320 loads against the 128 WMMAs freely.
// ---------------------------------------------------------------------------
struct Slice {
  v16bf a;
  v16bf b[4];
};

__device__ __forceinline__ Slice load_slice(const unsigned short* Ab,
                                            const unsigned short* Bb,
                                            int kk, int lane) {
  Slice s;
  s.a = load_frag(Ab + kk, 1024, lane);
#pragma unroll
  for (int t = 0; t < 4; ++t)
    s.b[t] = load_frag(Bb + (size_t)t * 16 * 1024 + kk, 1024, lane);
  return s;
}

__device__ __forceinline__ void gemm_k_loop(const unsigned short* Ab,
                                            const unsigned short* Bb,
                                            int lane, v8f acc[4]) {
  // 32 K-slices of depth 32 (K = 1024), software-pipelined depth 2.
  Slice s0 = load_slice(Ab, Bb, 0 * 32, lane);
  Slice s1 = load_slice(Ab, Bb, 1 * 32, lane);
#pragma unroll
  for (int i = 0; i < 15; ++i) {
#pragma unroll
    for (int t = 0; t < 4; ++t) acc[t] = WMMA_BF16(s0.a, s0.b[t], acc[t]);
    s0 = load_slice(Ab, Bb, (2 * i + 2) * 32, lane);
#pragma unroll
    for (int t = 0; t < 4; ++t) acc[t] = WMMA_BF16(s1.a, s1.b[t], acc[t]);
    s1 = load_slice(Ab, Bb, (2 * i + 3) * 32, lane);
  }
#pragma unroll
  for (int t = 0; t < 4; ++t) acc[t] = WMMA_BF16(s0.a, s0.b[t], acc[t]);
#pragma unroll
  for (int t = 0; t < 4; ++t) acc[t] = WMMA_BF16(s1.a, s1.b[t], acc[t]);
}

// ---------------------------------------------------------------------------
// GEMM1: qkv = x @ W_qkv + b_qkv -> Q[B,H,S,64], K[B,H,S,64], V^T[B,H,64,S].
// ---------------------------------------------------------------------------
__global__ void __launch_bounds__(256)
gemm_qkv_kernel(const unsigned short* __restrict__ xb,
                const unsigned short* __restrict__ wT,   // [3072][1024] N-major
                const float* __restrict__ bias,
                unsigned short* __restrict__ qb,
                unsigned short* __restrict__ kb,
                unsigned short* __restrict__ vT) {
  const int lane = threadIdx.x & 31;
  const int wid  = blockIdx.x * 8 + (threadIdx.x >> 5);  // 24576 waves
  const int nt   = wid % 48;
  const int mt   = wid / 48;                             // < 512

  const unsigned short* Ab = xb + (size_t)(mt * 16) * 1024;
  const unsigned short* Bb = wT + (size_t)(nt * 64) * 1024;

  v8f acc[4] = { {0,0,0,0,0,0,0,0}, {0,0,0,0,0,0,0,0},
                 {0,0,0,0,0,0,0,0}, {0,0,0,0,0,0,0,0} };
  gemm_k_loop(Ab, Bb, lane, acc);

  const int half8 = (lane >> 4) << 3;
  const int nlo   = lane & 15;
#pragma unroll
  for (int t = 0; t < 4; ++t) {
    const int n    = nt * 64 + t * 16 + nlo;
    const int hh   = n / 192;
    const int part = (n % 192) / 64;
    const int d    = n % 64;
    const float bn = bias[n];
#pragma unroll
    for (int j = 0; j < 8; ++j) {
      const int row = mt * 16 + j + half8;     // global token [0,8192)
      const int bb  = row >> 11;
      const int ss  = row & 2047;
      const unsigned short val = f2bf(acc[t][j] + bn);
      const size_t bh = (size_t)(bb * 16 + hh);
      if (part == 0)      qb[(bh * 2048 + ss) * 64 + d] = val;
      else if (part == 1) kb[(bh * 2048 + ss) * 64 + d] = val;
      else                vT[(bh * 64 + d) * 2048 + ss] = val;
    }
  }
}

// ---------------------------------------------------------------------------
// Flash attention. Block = 128 threads (4 waves) handling 64 consecutive
// queries of one (b,h); wave w owns 16 of them. Per 32-key block:
//   - K tile [32][64] and V^T tile [64][32] staged into LDS with
//     global_load_async_to_lds_b128, double-buffered.
//   - 4 WMMA for QK^T, online softmax (half-wave shfl_xor reductions),
//     P bounced through wave-private LDS (C-layout -> A-layout), 4 WMMA PV.
// ---------------------------------------------------------------------------
__global__ void __launch_bounds__(128)
attn_kernel(const unsigned short* __restrict__ qb,
            const unsigned short* __restrict__ kbuf,
            const unsigned short* __restrict__ vT,
            const float* __restrict__ mask,
            unsigned short* __restrict__ obf) {
  __shared__ __align__(16) unsigned short Kb[2][32 * 64];   // 2 x 4KB
  __shared__ __align__(16) unsigned short Vb[2][64 * 32];   // 2 x 4KB
  __shared__ __align__(16) unsigned short Pb[4][16 * 32];   // 4 x 1KB

  const int tid  = threadIdx.x;
  const int lane = tid & 31;
  const int w    = tid >> 5;
  const int bh   = blockIdx.x >> 5;       // [0,64): (b,h)
  const int qc   = blockIdx.x & 31;       // 64-query chunk, S/64 = 32
  const int bb   = bh >> 4;
  const int hh   = bh & 15;
  const int q0   = qc * 64 + w * 16;

  const unsigned short* Qp = qb   + (size_t)bh * 2048 * 64;
  const unsigned short* Kp = kbuf + (size_t)bh * 2048 * 64;
  const unsigned short* Vp = vT   + (size_t)bh * 64 * 2048;

  const v16bf qa0 = load_frag(Qp + (size_t)q0 * 64,      64, lane);
  const v16bf qa1 = load_frag(Qp + (size_t)q0 * 64 + 32, 64, lane);

  float mrow[8], lrow[8];
  v8f acc[4] = { {0,0,0,0,0,0,0,0}, {0,0,0,0,0,0,0,0},
                 {0,0,0,0,0,0,0,0}, {0,0,0,0,0,0,0,0} };
#pragma unroll
  for (int j = 0; j < 8; ++j) { mrow[j] = -3.0e38f; lrow[j] = 0.f; }

  const int half8 = (lane >> 4) << 3;
  const int nlo   = lane & 15;

  // --- cooperative async staging of one 32-key block -----------------------
  auto stage = [&](int k0, int buf) {
    // K tile: contiguous 4KB starting at Kp + k0*64 elements.
    {
      const char* g = (const char*)(Kp + (size_t)k0 * 64);
      const unsigned l = (unsigned)(uintptr_t)&Kb[buf][0];
      async_copy_b128(l + tid * 16,        g + tid * 16);
      async_copy_b128(l + 2048 + tid * 16, g + 2048 + tid * 16);
    }
    // V^T tile: 64 rows (d), 64 bytes each, at Vp + d*2048 + k0 elements.
    {
      const unsigned l = (unsigned)(uintptr_t)&Vb[buf][0];
      const int c0 = tid;           // chunk c: row c>>2, byte (c&3)*16
      const int r0 = c0 >> 2, o0 = (c0 & 3) * 16;
      async_copy_b128(l + c0 * 16, (const char*)(Vp + (size_t)r0 * 2048 + k0) + o0);
      const int c1 = tid + 128;
      const int r1 = c1 >> 2, o1 = (c1 & 3) * 16;
      async_copy_b128(l + c1 * 16, (const char*)(Vp + (size_t)r1 * 2048 + k0) + o1);
    }
  };

  stage(0, 0);

  for (int blk = 0; blk < 64; ++blk) {
    const int cur = blk & 1;
    const int k0  = blk * 32;

    wait_async0();        // my async copies into buf[cur] have landed
    __syncthreads();      // everyone's have; buf[cur] published

    if (blk + 1 < 64) stage((blk + 1) * 32, cur ^ 1);   // overlap with compute

    const unsigned short* Kl = &Kb[cur][0];
    const v16bf bk00 = load_frag(Kl,                64, lane);
    const v16bf bk01 = load_frag(Kl + 32,           64, lane);
    const v16bf bk10 = load_frag(Kl + 16 * 64,      64, lane);
    const v16bf bk11 = load_frag(Kl + 16 * 64 + 32, 64, lane);

    v8f c0 = {0,0,0,0,0,0,0,0};
    v8f c1 = {0,0,0,0,0,0,0,0};
    c0 = WMMA_BF16(qa0, bk00, c0);
    c0 = WMMA_BF16(qa1, bk01, c0);
    c1 = WMMA_BF16(qa0, bk10, c1);
    c1 = WMMA_BF16(qa1, bk11, c1);

    float alpha[8];
#pragma unroll
    for (int j = 0; j < 8; ++j) {
      const int qrow = q0 + j + half8;
      float s0 = c0[j] * 0.125f + mask[(size_t)qrow * 2048 + k0 + nlo];
      float s1 = c1[j] * 0.125f + mask[(size_t)qrow * 2048 + k0 + 16 + nlo];
      float bm = fmaxf(s0, s1);
#pragma unroll
      for (int off = 1; off < 16; off <<= 1)
        bm = fmaxf(bm, __shfl_xor(bm, off, 32));
      const float nm = fmaxf(mrow[j], bm);
      const float p0 = __expf(s0 - nm);
      const float p1 = __expf(s1 - nm);
      float rs = p0 + p1;
#pragma unroll
      for (int off = 1; off < 16; off <<= 1)
        rs += __shfl_xor(rs, off, 32);
      alpha[j] = __expf(mrow[j] - nm);
      lrow[j]  = lrow[j] * alpha[j] + rs;
      mrow[j]  = nm;
      // C-layout -> LDS row-major P[16][32] (wave-private region)
      const int m = j + half8;
      Pb[w][m * 32 + nlo]      = f2bf(p0);
      Pb[w][m * 32 + 16 + nlo] = f2bf(p1);
    }

#pragma unroll
    for (int t = 0; t < 4; ++t)
#pragma unroll
      for (int j = 0; j < 8; ++j)
        acc[t][j] *= alpha[j];

    // DS ops from one wave execute in order: loads below observe the stores
    // above without a barrier (region is wave-private).
    const v16bf pa = load_frag(&Pb[w][0], 32, lane);

    const unsigned short* Vl = &Vb[cur][0];
#pragma unroll
    for (int t = 0; t < 4; ++t) {
      const v16bf bv = load_frag(Vl + t * 16 * 32, 32, lane);
      acc[t] = WMMA_BF16(pa, bv, acc[t]);
    }

    __syncthreads();      // all waves done reading buf[cur] before it refills
  }

  // O /= l, write [B,S,D] bf16 for the output projection
#pragma unroll
  for (int t = 0; t < 4; ++t) {
#pragma unroll
    for (int j = 0; j < 8; ++j) {
      const int qrow = q0 + j + half8;
      const int d    = t * 16 + nlo;
      const float v  = acc[t][j] / lrow[j];
      obf[((size_t)bb * 2048 + qrow) * 1024 + hh * 64 + d] = f2bf(v);
    }
  }
}

// ---------------------------------------------------------------------------
// GEMM2: out = vals @ W_out + b_out (fp32).
// ---------------------------------------------------------------------------
__global__ void __launch_bounds__(256)
gemm_out_kernel(const unsigned short* __restrict__ ob,
                const unsigned short* __restrict__ wT,   // [1024][1024] N-major
                const float* __restrict__ bias,
                float* __restrict__ out) {
  const int lane = threadIdx.x & 31;
  const int wid  = blockIdx.x * 8 + (threadIdx.x >> 5);  // 8192 waves
  const int nt   = wid & 15;
  const int mt   = wid >> 4;                             // < 512

  const unsigned short* Ab = ob + (size_t)(mt * 16) * 1024;
  const unsigned short* Bb = wT + (size_t)(nt * 64) * 1024;

  v8f acc[4] = { {0,0,0,0,0,0,0,0}, {0,0,0,0,0,0,0,0},
                 {0,0,0,0,0,0,0,0}, {0,0,0,0,0,0,0,0} };
  gemm_k_loop(Ab, Bb, lane, acc);

  const int half8 = (lane >> 4) << 3;
  const int nlo   = lane & 15;
#pragma unroll
  for (int t = 0; t < 4; ++t) {
    const int n    = nt * 64 + t * 16 + nlo;
    const float bn = bias[n];
#pragma unroll
    for (int j = 0; j < 8; ++j) {
      const int row = mt * 16 + j + half8;
      out[(size_t)row * 1024 + n] = acc[t][j] + bn;
    }
  }
}

// ---------------------------------------------------------------------------
// Host launcher. Workspace layout (bytes):
//   xb   @ 0         : 16 MiB  x as bf16            [8192 x 1024]
//   wqT  @ 16777216  :  6 MiB  W_qkv^T bf16         [3072 x 1024]
//   woT  @ 23068672  :  2 MiB  W_out^T bf16         [1024 x 1024]
//   qb   @ 25165824  : 16 MiB  Q  [B,H,S,64] bf16
//   kb   @ 41943040  : 16 MiB  K  [B,H,S,64] bf16
//   vT   @ 58720256  : 16 MiB  V^T[B,H,64,S] bf16
//   ob   @ 75497472  : 16 MiB  attn output [B,S,D] bf16
// total ~88 MiB.
// ---------------------------------------------------------------------------
extern "C" void kernel_launch(void* const* d_in, const int* in_sizes, int n_in,
                              void* d_out, int out_size, void* d_ws, size_t ws_size,
                              hipStream_t stream) {
  const float* x    = (const float*)d_in[0];
  const float* mask = (const float*)d_in[1];
  const float* Wqkv = (const float*)d_in[2];
  const float* bqkv = (const float*)d_in[3];
  const float* Wout = (const float*)d_in[4];
  const float* bout = (const float*)d_in[5];
  float* out = (float*)d_out;

  char* ws = (char*)d_ws;
  unsigned short* xb  = (unsigned short*)(ws + 0);
  unsigned short* wqT = (unsigned short*)(ws + 16777216);
  unsigned short* woT = (unsigned short*)(ws + 23068672);
  unsigned short* qb  = (unsigned short*)(ws + 25165824);
  unsigned short* kb  = (unsigned short*)(ws + 41943040);
  unsigned short* vT  = (unsigned short*)(ws + 58720256);
  unsigned short* ob  = (unsigned short*)(ws + 75497472);

  f32_to_bf16x4_kernel<<<8192, 256, 0, stream>>>(x, xb, 2097152);
  transpose_bf16_kernel<<<dim3(32, 96), 256, 0, stream>>>(Wqkv, wqT, 3072, 1024);
  transpose_bf16_kernel<<<dim3(32, 32), 256, 0, stream>>>(Wout, woT, 1024, 1024);

  gemm_qkv_kernel<<<3072, 256, 0, stream>>>(xb, wqT, bqkv, qb, kb, vT);
  attn_kernel<<<2048, 128, 0, stream>>>(qb, kb, vT, mask, ob);
  gemm_out_kernel<<<1024, 256, 0, stream>>>(ob, woT, bout, out);
}
